// RNN_77635828843197
// MI455X (gfx1250) — matrix operand order repaired and checked by reading, not compiled, not run on hardware
//
#include <hip/hip_runtime.h>
#include <hip/hip_bf16.h>
#include <math.h>

typedef __bf16 bf16;
typedef __attribute__((ext_vector_type(16))) __bf16 v16bf;
typedef __attribute__((ext_vector_type(8)))  __bf16 v8bf;
typedef __attribute__((ext_vector_type(8)))  float  v8f;

#define DEVINL __device__ __forceinline__

constexpr int Bv = 16, Tv = 128, Sv = 1024, Ev = 512, Vv = 32000, Lv = 2;
constexpr int Mv = Bv * Tv;   // 2048 rows, row m = t*B + b

// ---- WMMA wrapper (guarded so the host pass never sees the amdgcn builtin) ----
DEVINL v8f wmma_bf16(v16bf a, v16bf b, v8f c) {
#if defined(__AMDGCN__)
  // emits v_wmma_f32_16x16x32_bf16
  return __builtin_amdgcn_wmma_f32_16x16x32_bf16(false, a, false, b, (short)0, c,
                                                 false, false);
#else
  (void)a; (void)b;
  return c;
#endif
}

// A fragment: 16x32 bf16, row-major source, row m = lane&15.
// Order used here (same as load_b): per lane 16 contiguous k at k-offset kh*16.
DEVINL v16bf load_a(const bf16* __restrict__ base, int lda, int k0) {
  const int lane = threadIdx.x & 31;
  const int m = lane & 15, kh = lane >> 4;
  const bf16* p = base + (size_t)m * lda + k0 + kh * 16;
  v8bf c0 = *(const v8bf*)p;
  v8bf c1 = *(const v8bf*)(p + 8);
  v16bf a;
#pragma unroll
  for (int i = 0; i < 8; ++i) { a[i] = c0[i]; a[i + 8] = c1[i]; }
  return a;
}

// B fragment: B[k][n] = W[n][k] (W row-major, ld = ldw). Column n = contiguous W row.
// Order used here (same as load_a): lanes 0-15 hold K=0..15, lanes 16-31 hold K=16..31 (contiguous 32B).
DEVINL v16bf load_b(const bf16* __restrict__ wbase, int ldw, int k0) {
  const int lane = threadIdx.x & 31;
  const int n = lane & 15, kh = lane >> 4;
  const bf16* p = wbase + (size_t)n * ldw + k0 + kh * 16;
  v8bf c0 = *(const v8bf*)p;
  v8bf c1 = *(const v8bf*)(p + 8);
  v16bf b;
#pragma unroll
  for (int i = 0; i < 8; ++i) { b[i] = c0[i]; b[i + 8] = c1[i]; }
  return b;
}

// Fallback: B fragment converted from f32 weights in-register.
DEVINL v16bf load_b_f32(const float* __restrict__ wbase, int ldw, int k0) {
  const int lane = threadIdx.x & 31;
  const int n = lane & 15, kh = lane >> 4;
  const float* p = wbase + (size_t)n * ldw + k0 + kh * 16;
  v16bf b;
#pragma unroll
  for (int i = 0; i < 16; i += 4) {
    float4 f = *(const float4*)(p + i);
    b[i] = (bf16)f.x; b[i + 1] = (bf16)f.y; b[i + 2] = (bf16)f.z; b[i + 3] = (bf16)f.w;
  }
  return b;
}

// ---------------- utility kernels ----------------
__global__ void k_cvt_bf16(const float* __restrict__ src, bf16* __restrict__ dst,
                           long n) {
  long i = (long)blockIdx.x * blockDim.x + threadIdx.x;
  long stride = (long)gridDim.x * blockDim.x;
  for (; i < n; i += stride) dst[i] = (bf16)src[i];
}

__global__ void k_zero_bf16(bf16* __restrict__ dst, int n) {
  int i = blockIdx.x * blockDim.x + threadIdx.x;
  if (i < n) dst[i] = (bf16)0.0f;
}

// Xb[m][e] = bf16(emb[tokens[b][t]][e]),  m = t*B + b
__global__ void k_gather_x(const int* __restrict__ tokens,
                           const float* __restrict__ emb,
                           bf16* __restrict__ Xb) {
  const int m = blockIdx.x;
  const int b = m & (Bv - 1), t = m >> 4;
  const int tok = tokens[b * Tv + t];
  const float* src = emb + (size_t)tok * Ev;
  bf16* dst = Xb + (size_t)m * Ev;
  for (int e = threadIdx.x; e < Ev; e += blockDim.x) dst[e] = (bf16)src[e];
}

// ---------------- batched input projection: Xw[l][m][s] = Xb[m] . Wx[l][s] ----------------
__global__ void __launch_bounds__(256)
k_xw_gemm(const bf16* __restrict__ Xb, const bf16* __restrict__ Wxb,
          float* __restrict__ Xw) {
  const int l = blockIdx.y;
  const int w = threadIdx.x >> 5;
  const int gw = blockIdx.x * 8 + w;     // 0..8191 = 128 Mtiles * 64 Ntiles
  const int mtile = gw & 127;
  const int ntile = gw >> 7;
  const bf16* abase = Xb + (size_t)mtile * 16 * Ev;
  const bf16* bbase = Wxb + (size_t)l * Sv * Ev + (size_t)ntile * 16 * Ev;
  v8f acc = {};
#pragma unroll 4
  for (int k0 = 0; k0 < Ev; k0 += 32) {
    v16bf a = load_a(abase, Ev, k0);
    v16bf bb = load_b(bbase, Ev, k0);
    acc = wmma_bf16(a, bb, acc);
  }
  const int lane = threadIdx.x & 31;
  const int nn = lane & 15, kh = lane >> 4;
  float* out = Xw + (size_t)l * Mv * Sv;
#pragma unroll
  for (int r = 0; r < 8; ++r) {
    const int rr = kh * 8 + r;
    out[(size_t)(mtile * 16 + rr) * Sv + ntile * 16 + nn] = acc[r];
  }
}

// ---------------- one recurrence step, one layer: h' = tanh(LN(xw + h @ Wh^T)) ----------------
// Single workgroup (1024 thr = 32 waves); wave w owns N-tiles 2w, 2w+1.
__global__ void __launch_bounds__(1024)
k_rnn_step(const bf16* __restrict__ hin, const bf16* __restrict__ Whl,
           const float* __restrict__ xw,                   // (B,S) slice of Xw
           const float* __restrict__ g, const float* __restrict__ bt,
           bf16* __restrict__ hout, bf16* __restrict__ hall) {
  __shared__ float s_a[Bv][Sv + 8];      // 16x1024 f32 (+pad) = 64KB, fits 320KB LDS
  __shared__ float s_mu[Bv], s_rs[Bv];
  const int w = threadIdx.x >> 5;
  const int lane = threadIdx.x & 31;
  const int nn = lane & 15, kh = lane >> 4;
  const int nt0 = 2 * w, nt1 = 2 * w + 1;
  const bf16* b0base = Whl + (size_t)nt0 * 16 * Sv;
  const bf16* b1base = Whl + (size_t)nt1 * 16 * Sv;
  v8f acc0 = {}, acc1 = {};
  for (int k0 = 0; k0 < Sv; k0 += 32) {   // 32 iters, A fragment shared by both tiles
    v16bf a = load_a(hin, Sv, k0);
    v16bf bb0 = load_b(b0base, Sv, k0);
    v16bf bb1 = load_b(b1base, Sv, k0);
    acc0 = wmma_bf16(a, bb0, acc0);
    acc1 = wmma_bf16(a, bb1, acc1);
  }
#pragma unroll
  for (int r = 0; r < 8; ++r) {
    const int rr = kh * 8 + r;
    const int c0 = nt0 * 16 + nn, c1 = nt1 * 16 + nn;
    s_a[rr][c0] = acc0[r] + xw[(size_t)rr * Sv + c0];
    s_a[rr][c1] = acc1[r] + xw[(size_t)rr * Sv + c1];
  }
  __syncthreads();
  if (w < Bv) {                           // waves 0..15 each reduce one row (wave32 shuffles)
    float s = 0.f, ss = 0.f;
    for (int j = 0; j < Sv / 32; ++j) {
      float v = s_a[w][lane + 32 * j];
      s += v; ss += v * v;
    }
#pragma unroll
    for (int off = 16; off > 0; off >>= 1) {
      s += __shfl_xor(s, off);
      ss += __shfl_xor(ss, off);
    }
    if (lane == 0) {
      const float mu = s * (1.0f / Sv);
      const float var = ss * (1.0f / Sv) - mu * mu;
      s_mu[w] = mu;
      s_rs[w] = rsqrtf(var + 1e-5f);
    }
  }
  __syncthreads();
  for (int idx = threadIdx.x; idx < Bv * Sv; idx += blockDim.x) {
    const int row = idx >> 10, col = idx & (Sv - 1);
    const float y = tanhf((s_a[row][col] - s_mu[row]) * s_rs[row] * g[col] + bt[col]);
    const bf16 hb = (bf16)y;
    hout[(size_t)row * Sv + col] = hb;
    if (hall) hall[(size_t)row * Sv + col] = hb;   // stash h_t for the batched logits GEMM
  }
}

// ---------------- batched logits: out[b][t][v] = Hb[t*16+b] . Wout[v] + bout[v] ----------------
// 4 M-tiles x 4 N-tiles per wave: 16 accumulators (128 VGPRs), 4 A + 4 B fragment
// loads feed 16 WMMAs per K-step; Wout traversed only 32x (vs 128x with 1 M-tile).
// Grid: 32 Mgroups * 500 Ngroups = 16000 waves = 2000 WGs * 8 waves (exact -> EXEC all 1s).
template <bool WB16>
__global__ void __launch_bounds__(256)
k_logits_t(const bf16* __restrict__ Hb, const bf16* __restrict__ Woutb,
           const float* __restrict__ Woutf, const float* __restrict__ bout,
           float* __restrict__ out) {
  const int w = threadIdx.x >> 5;
  const int gw = blockIdx.x * 8 + w;      // 0..15999
  const int mg = gw & 31;                 // M group: rows of 4 consecutive t-tiles
  const int ng = gw >> 5;                 // 0..499 -> 64 vocab columns
  const int cb = ng * 64;

  const bf16* ab[4];
#pragma unroll
  for (int i = 0; i < 4; ++i) ab[i] = Hb + (size_t)(mg * 4 + i) * 16 * Sv;

  v8f acc[4][4];
#pragma unroll
  for (int i = 0; i < 4; ++i)
#pragma unroll
    for (int j = 0; j < 4; ++j) acc[i][j] = v8f{};

  for (int k0 = 0; k0 < Sv; k0 += 32) {
    v16bf a0 = load_a(ab[0], Sv, k0);
    v16bf a1 = load_a(ab[1], Sv, k0);
    v16bf a2 = load_a(ab[2], Sv, k0);
    v16bf a3 = load_a(ab[3], Sv, k0);
#pragma unroll
    for (int j = 0; j < 4; ++j) {
      v16bf bb;
      if constexpr (WB16)
        bb = load_b(Woutb + (size_t)(cb + j * 16) * Sv, Sv, k0);
      else
        bb = load_b_f32(Woutf + (size_t)(cb + j * 16) * Sv, Sv, k0);
      acc[0][j] = wmma_bf16(a0, bb, acc[0][j]);
      acc[1][j] = wmma_bf16(a1, bb, acc[1][j]);
      acc[2][j] = wmma_bf16(a2, bb, acc[2][j]);
      acc[3][j] = wmma_bf16(a3, bb, acc[3][j]);
    }
  }

  const int lane = threadIdx.x & 31;
  const int nn = lane & 15, kh = lane >> 4;
#pragma unroll
  for (int j = 0; j < 4; ++j) {
    const int n = cb + j * 16 + nn;
    const float bo = bout[n];
#pragma unroll
    for (int i = 0; i < 4; ++i) {
      const int mtile = mg * 4 + i;       // == t
#pragma unroll
      for (int r = 0; r < 8; ++r) {
        const int rr = kh * 8 + r;        // == b
        out[((size_t)rr * Tv + mtile) * Vv + n] = acc[i][j][r] + bo;
      }
    }
  }
}

// ---------------- host side ----------------
extern "C" void kernel_launch(void* const* d_in, const int* in_sizes, int n_in,
                              void* d_out, int out_size, void* d_ws, size_t ws_size,
                              hipStream_t stream) {
  (void)in_sizes; (void)n_in; (void)out_size;
  const int*   tokens = (const int*)d_in[0];
  const float* emb    = (const float*)d_in[1];
  const float* Wx     = (const float*)d_in[2];
  const float* Wh     = (const float*)d_in[3];
  const float* gamma  = (const float*)d_in[4];
  const float* beta   = (const float*)d_in[5];
  const float* Wout   = (const float*)d_in[6];
  const float* bout   = (const float*)d_in[7];
  float* out = (float*)d_out;

  // workspace layout (all 256B aligned); base ~29.4 MB, +65.5 MB for bf16 Wout
  char* ws = (char*)d_ws;
  bf16*  Xb  = (bf16*)(ws + 0);           //  2 MB : (2048,512) bf16
  bf16*  Wxb = (bf16*)(ws + 2097152);     //  2 MB : (2,1024,512) bf16
  bf16*  Whb = (bf16*)(ws + 4194304);     //  4 MB : (2,1024,1024) bf16
  float* Xw  = (float*)(ws + 8388608);    // 16 MB : (2,2048,1024) f32
  bf16*  Hb  = (bf16*)(ws + 25165824);    //  4 MB : (2048,1024) bf16
  bf16*  hA  = (bf16*)(ws + 29360128);    // 32 KB : (16,1024) bf16
  bf16*  hB  = (bf16*)(ws + 29392896);    // 32 KB
  const size_t WOUTB_OFF = 29425664;      // 65.5 MB : (32000,1024) bf16
  bf16*  Woutb = (bf16*)(ws + WOUTB_OFF);
  const bool use_wb16 = ws_size >= WOUTB_OFF + (size_t)Vv * Sv * sizeof(bf16);

  k_cvt_bf16<<<2048, 256, 0, stream>>>(Wx, Wxb, (long)Lv * Sv * Ev);
  k_cvt_bf16<<<2048, 256, 0, stream>>>(Wh, Whb, (long)Lv * Sv * Sv);
  if (use_wb16)
    k_cvt_bf16<<<8192, 256, 0, stream>>>(Wout, Woutb, (long)Vv * Sv);
  k_gather_x<<<Mv, 256, 0, stream>>>(tokens, emb, Xb);
  k_xw_gemm<<<dim3(1024, 2), 256, 0, stream>>>(Xb, Wxb, Xw);
  k_zero_bf16<<<(Bv * Sv + 255) / 256, 256, 0, stream>>>(hA, Bv * Sv);  // h0 = 0

  for (int t = 0; t < Tv; ++t) {
    k_rnn_step<<<1, 1024, 0, stream>>>(
        hA, Whb, Xw + ((size_t)0 * Mv + (size_t)t * Bv) * Sv,
        gamma, beta, hB, (bf16*)nullptr);
    k_rnn_step<<<1, 1024, 0, stream>>>(
        hB, Whb + (size_t)Sv * Sv, Xw + ((size_t)1 * Mv + (size_t)t * Bv) * Sv,
        gamma + Sv, beta + Sv, hA, Hb + (size_t)t * Bv * Sv);
  }

  if (use_wb16)
    k_logits_t<true><<<2000, 256, 0, stream>>>(Hb, Woutb, nullptr, bout, out);
  else
    k_logits_t<false><<<2000, 256, 0, stream>>>(Hb, nullptr, Wout, bout, out);
}
